// MultiQueryAttention_50594714747435
// MI455X (gfx1250) — matrix operand order repaired and checked by reading, not compiled, not run on hardware
//
#include <hip/hip_runtime.h>

// ---------------- problem constants (from reference) ----------------
#define SEQ    2048
#define NBATCH 2
#define DMODEL 2048
#define NQH    16
#define NKVH   4
#define HEADD  128
#define QK_SCALE 0.08838834764831845f   // 1/sqrt(128)

typedef __attribute__((ext_vector_type(16))) _Float16 v16h;
typedef __attribute__((ext_vector_type(8)))  _Float16 v8h;
typedef __attribute__((ext_vector_type(4)))  _Float16 v4h;
typedef __attribute__((ext_vector_type(8)))  float    v8f;

union V16U { v16h v; v8h h[2]; };

// D = A(16x32 f16) x B(32x16 f16) + C(16x16 f32)
__device__ __forceinline__ v8f wmma_f16f32(v16h a, v16h b, v8f c) {
  return __builtin_amdgcn_wmma_f32_16x16x32_f16(false, a, false, b, (short)0, c,
                                                false, false);
}

// A fragment: A row-major [M][K], lane r holds row M=Mbase+r,
// halves K = Kbase + 8*hi + h  and  Kbase + 16 + 8*hi + h   (ISA 7.12.2)
__device__ __forceinline__ v16h load_frag_a(const _Float16* __restrict__ A, int lda,
                                            int Mbase, int Kbase, int lane) {
  int r = lane & 15, hi = lane >> 4;
  const _Float16* p = A + (size_t)(Mbase + r) * lda + Kbase + 8 * hi;
  V16U u;
  u.h[0] = *(const v8h*)p;
  u.h[1] = *(const v8h*)(p + 16);
  return u.v;
}

// B fragment from B^T row-major [N][K]: lane n holds column N=Nbase+n,
// halves K = Kbase + 8*hi + h  and  Kbase + 16 + 8*hi + h
__device__ __forceinline__ v16h load_frag_bt(const _Float16* __restrict__ BT, int ldbt,
                                             int Kbase, int Nbase, int lane) {
  int n = lane & 15, hi = lane >> 4;
  const _Float16* p = BT + (size_t)(Nbase + n) * ldbt + Kbase + 8 * hi;
  V16U u;
  u.h[0] = *(const v8h*)p;
  u.h[1] = *(const v8h*)(p + 16);
  return u.v;
}

// ---------------- f32 -> f16 converters ----------------
__global__ __launch_bounds__(256)
void mqa_cvt_kernel(const float* __restrict__ in, _Float16* __restrict__ out, int n) {
  int idx = (blockIdx.x * blockDim.x + threadIdx.x) * 4;
  if (idx < n) {
    float4 f = *(const float4*)(in + idx);
    v4h o = {(_Float16)f.x, (_Float16)f.y, (_Float16)f.z, (_Float16)f.w};
    *(v4h*)(out + idx) = o;
  }
}

// in: [K][N] f32 row-major  ->  out: [N][K] f16 row-major (transpose)
__global__ __launch_bounds__(256)
void mqa_cvtT_kernel(const float* __restrict__ in, _Float16* __restrict__ out,
                     int K, int N) {
  int i = blockIdx.x * blockDim.x + threadIdx.x;
  if (i < K * N) {
    int nn = i / K;
    int kk = i - nn * K;
    out[i] = (_Float16)in[(size_t)kk * N + nn];
  }
}

// ---------------- generic WMMA GEMM: C[M][N] = A[M][K] * B^T[N][K] ----------------
// Block: 8 waves as 2(M) x 4(N); block tile 64M x 256N; wave tile 32M x 64N.
// mode 0: write f32 to out_f row-major [M][N]                (final out-proj)
// mode 1: RoPE + *QK_SCALE, f16 -> out_h[b][head][s][d]      (Q)
// mode 2: RoPE,             f16 -> out_h[b][head][s][d]      (K)
// mode 3: plain,            f16 -> out_h[b][head][d][s]      (V transposed)
__global__ __launch_bounds__(256)
void mqa_proj_kernel(const _Float16* __restrict__ A, const _Float16* __restrict__ BT,
                     int M, int N, int K,
                     const float* __restrict__ cosp, const float* __restrict__ sinp,
                     _Float16* __restrict__ out_h, float* __restrict__ out_f,
                     int mode, int nheads) {
  int lane = threadIdx.x & 31;
  int wave = threadIdx.x >> 5;
  int wm = wave & 1, wn = wave >> 1;
  int Mb = blockIdx.y * 64 + wm * 32;
  int Nb = blockIdx.x * 256 + wn * 64;
  int r = lane & 15, hi = lane >> 4;

  const v8f vzero = {0.f, 0.f, 0.f, 0.f, 0.f, 0.f, 0.f, 0.f};
  v8f acc[2][4];
#pragma unroll
  for (int mi = 0; mi < 2; ++mi)
#pragma unroll
    for (int t = 0; t < 4; ++t) acc[mi][t] = vzero;

  for (int k = 0; k < K; k += 32) {
    v16h af0 = load_frag_a(A, K, Mb, k, lane);
    v16h af1 = load_frag_a(A, K, Mb + 16, k, lane);
#pragma unroll
    for (int t = 0; t < 4; ++t) {
      v16h bf = load_frag_bt(BT, K, k, Nb + 16 * t, lane);
      acc[0][t] = wmma_f16f32(af0, bf, acc[0][t]);
      acc[1][t] = wmma_f16f32(af1, bf, acc[1][t]);
    }
  }

  if (mode == 0) {
#pragma unroll
    for (int mi = 0; mi < 2; ++mi)
#pragma unroll
      for (int t = 0; t < 4; ++t) {
        int n = Nb + 16 * t + r;
#pragma unroll
        for (int j = 0; j < 8; ++j) {
          int row = Mb + 16 * mi + j + 8 * hi;
          out_f[(size_t)row * N + n] = acc[mi][t][j];
        }
      }
  } else if (mode == 3) {
    // V^T: out_h[((b*nheads+head)*HEADD + d)*SEQ + s]; 8 consecutive s -> one 16B store
#pragma unroll
    for (int mi = 0; mi < 2; ++mi)
#pragma unroll
      for (int t = 0; t < 4; ++t) {
        int n = Nb + 16 * t + r;
        int head = n >> 7, d = n & (HEADD - 1);
        v8h pk;
#pragma unroll
        for (int j = 0; j < 8; ++j) pk[j] = (_Float16)acc[mi][t][j];
        int row0 = Mb + 16 * mi + 8 * hi;
        int s0 = row0 & (SEQ - 1), bI = row0 >> 11;
        *(v8h*)(out_h + (((size_t)bI * nheads + head) * HEADD + d) * SEQ + s0) = pk;
      }
  } else {
    // RoPE: even d: x*c - x[d+1]*s ; odd d: x*c + x[d-1]*s. Pair lives in lane^1.
    float sc = (mode == 1) ? QK_SCALE : 1.0f;
#pragma unroll
    for (int mi = 0; mi < 2; ++mi)
#pragma unroll
      for (int t = 0; t < 4; ++t) {
        int n = Nb + 16 * t + r;
        int head = n >> 7, d = n & (HEADD - 1);
        float sgn = (d & 1) ? 1.0f : -1.0f;
#pragma unroll
        for (int j = 0; j < 8; ++j) {
          int row = Mb + 16 * mi + j + 8 * hi;
          int s = row & (SEQ - 1), bI = row >> 11;
          float c  = cosp[s * HEADD + d];
          float sv = sinp[s * HEADD + d];
          float val = acc[mi][t][j];
          float partner = __shfl_xor(val, 1, 32);
          float res = (val * c + sgn * partner * sv) * sc;
          out_h[(((size_t)bI * nheads + head) * SEQ + s) * HEADD + d] = (_Float16)res;
        }
      }
  }
}

// ---------------- flash attention: one wave = 32 query rows ----------------
// qh : [B][NQH][S][HEADD] f16 (pre-scaled by QK_SCALE)
// kh : [B][NKVH][S][HEADD] f16 (B^T for logits: N=key, K=d)
// vT : [B][NKVH][HEADD][S] f16 (B^T for PV:     N=d,   K=key)
// oh : [B][S][NQH*HEADD]   f16
__global__ __launch_bounds__(256)
void mqa_attn_kernel(const _Float16* __restrict__ qh, const _Float16* __restrict__ kh,
                     const _Float16* __restrict__ vT, _Float16* __restrict__ oh) {
  __shared__ __align__(32) _Float16 plds[8][2][16][32];

  int lane = threadIdx.x & 31;
  int wave = threadIdx.x >> 5;
  int r = lane & 15, hi = lane >> 4;
  int h = blockIdx.y, b = blockIdx.z, hv = h & (NKVH - 1);
  int q0 = blockIdx.x * 256 + wave * 32;

  const _Float16* qbase = qh + (size_t)(b * NQH + h) * SEQ * HEADD;
  const _Float16* kbase = kh + (size_t)(b * NKVH + hv) * SEQ * HEADD;
  const _Float16* vbase = vT + (size_t)(b * NKVH + hv) * HEADD * SEQ;

  // Preload the wave's 32x128 Q tile as 2x4 A fragments (reused every chunk)
  v16h qf[2][4];
#pragma unroll
  for (int mi = 0; mi < 2; ++mi)
#pragma unroll
    for (int kd = 0; kd < 4; ++kd)
      qf[mi][kd] = load_frag_a(qbase, HEADD, q0 + 16 * mi, 32 * kd, lane);

  const v8f vzero = {0.f, 0.f, 0.f, 0.f, 0.f, 0.f, 0.f, 0.f};
  v8f O[2][8];
  float m_i[2][8], l_i[2][8];
#pragma unroll
  for (int mi = 0; mi < 2; ++mi) {
#pragma unroll
    for (int t = 0; t < 8; ++t) O[mi][t] = vzero;
#pragma unroll
    for (int j = 0; j < 8; ++j) { m_i[mi][j] = -3.0e38f; l_i[mi][j] = 0.0f; }
  }

  for (int kb = 0; kb < SEQ; kb += 32) {
    // logits tile: 32 q-rows x 32 keys (K/V fragments reused across both mi)
    v8f s0[2] = {vzero, vzero}, s1[2] = {vzero, vzero};
#pragma unroll
    for (int kd = 0; kd < 4; ++kd) {
      v16h b0 = load_frag_bt(kbase, HEADD, 32 * kd, kb, lane);
      v16h b1 = load_frag_bt(kbase, HEADD, 32 * kd, kb + 16, lane);
#pragma unroll
      for (int mi = 0; mi < 2; ++mi) {
        s0[mi] = wmma_f16f32(qf[mi][kd], b0, s0[mi]);
        s1[mi] = wmma_f16f32(qf[mi][kd], b1, s1[mi]);
      }
    }

    // online softmax (row M = j + 8*hi; N across the 16 lanes of each half)
#pragma unroll
    for (int mi = 0; mi < 2; ++mi) {
#pragma unroll
      for (int j = 0; j < 8; ++j) {
        float mx = fmaxf(s0[mi][j], s1[mi][j]);
#pragma unroll
        for (int o = 1; o < 16; o <<= 1) mx = fmaxf(mx, __shfl_xor(mx, o, 32));
        float mnew = fmaxf(m_i[mi][j], mx);
        float alpha = __expf(m_i[mi][j] - mnew);
        float p0 = __expf(s0[mi][j] - mnew);
        float p1 = __expf(s1[mi][j] - mnew);
        float rs = p0 + p1;
#pragma unroll
        for (int o = 1; o < 16; o <<= 1) rs += __shfl_xor(rs, o, 32);
        l_i[mi][j] = l_i[mi][j] * alpha + rs;
        m_i[mi][j] = mnew;
#pragma unroll
        for (int t = 0; t < 8; ++t) O[mi][t][j] *= alpha;
        // stage P (D-layout) into LDS for A-layout reload
        plds[wave][mi][j + 8 * hi][r]      = (_Float16)p0;
        plds[wave][mi][j + 8 * hi][r + 16] = (_Float16)p1;
      }
    }

    // reload P as 16x32 A fragments (same-wave DS ops are in-order)
    V16U pu[2];
#pragma unroll
    for (int mi = 0; mi < 2; ++mi) {
      pu[mi].h[0] = *(const v8h*)&plds[wave][mi][r][8 * hi];
      pu[mi].h[1] = *(const v8h*)&plds[wave][mi][r][16 + 8 * hi];
    }

    // O += P @ V  (K = 32 keys, N = 128 d in 8 tiles; V fragment reused 2x)
#pragma unroll
    for (int t = 0; t < 8; ++t) {
      v16h bv = load_frag_bt(vbase, SEQ, kb, 16 * t, lane);
#pragma unroll
      for (int mi = 0; mi < 2; ++mi) O[mi][t] = wmma_f16f32(pu[mi].v, bv, O[mi][t]);
    }
  }

  // normalize and store: oh[b][s][h*128 + d]
#pragma unroll
  for (int mi = 0; mi < 2; ++mi)
#pragma unroll
    for (int j = 0; j < 8; ++j) {
      float inv = 1.0f / l_i[mi][j];
      int s = q0 + 16 * mi + j + 8 * hi;
#pragma unroll
      for (int t = 0; t < 8; ++t) {
        float val = O[mi][t][j] * inv;
        oh[(size_t)(b * SEQ + s) * (NQH * HEADD) + h * HEADD + 16 * t + r] = (_Float16)val;
      }
    }
}

// ---------------- host launcher ----------------
extern "C" void kernel_launch(void* const* d_in, const int* in_sizes, int n_in,
                              void* d_out, int out_size, void* d_ws, size_t ws_size,
                              hipStream_t stream) {
  const float* x    = (const float*)d_in[0];
  const float* cosp = (const float*)d_in[1];
  const float* sinp = (const float*)d_in[2];
  const float* Wq   = (const float*)d_in[3];
  const float* Wk   = (const float*)d_in[4];
  const float* Wv   = (const float*)d_in[5];
  const float* Wo   = (const float*)d_in[6];
  float* out = (float*)d_out;

  const size_t nX  = (size_t)NBATCH * SEQ * DMODEL;       // 8,388,608
  const size_t nWq = (size_t)DMODEL * NQH * HEADD;        // 4,194,304
  const size_t nWk = (size_t)DMODEL * NKVH * HEADD;       // 1,048,576
  const size_t nQ  = (size_t)NBATCH * NQH * SEQ * HEADD;  // 8,388,608
  const size_t nK  = (size_t)NBATCH * NKVH * SEQ * HEADD; // 2,097,152

  char* w = (char*)d_ws;
  auto alloc = [&](size_t elems) {
    _Float16* p = (_Float16*)w;
    w += (elems * sizeof(_Float16) + 255) & ~(size_t)255;
    return p;
  };
  _Float16* x_h    = alloc(nX);
  _Float16* Wq_t   = alloc(nWq);
  _Float16* Wk_t   = alloc(nWk);
  _Float16* Wv_t   = alloc(nWk);
  _Float16* Wo_t   = alloc(nWq);
  _Float16* q_h    = alloc(nQ);
  _Float16* k_h    = alloc(nK);
  _Float16* vT_h   = alloc(nK);
  _Float16* attn_h = alloc(nX);

  // stage 1: precision conversion (+ weight transposes for B^T fragment loads)
  mqa_cvt_kernel<<<dim3((unsigned)(nX / 4 / 256)), 256, 0, stream>>>(x, x_h, (int)nX);
  mqa_cvtT_kernel<<<dim3((unsigned)((nWq + 255) / 256)), 256, 0, stream>>>(Wq, Wq_t, DMODEL, NQH * HEADD);
  mqa_cvtT_kernel<<<dim3((unsigned)((nWk + 255) / 256)), 256, 0, stream>>>(Wk, Wk_t, DMODEL, NKVH * HEADD);
  mqa_cvtT_kernel<<<dim3((unsigned)((nWk + 255) / 256)), 256, 0, stream>>>(Wv, Wv_t, DMODEL, NKVH * HEADD);
  mqa_cvtT_kernel<<<dim3((unsigned)((nWq + 255) / 256)), 256, 0, stream>>>(Wo, Wo_t, NQH * HEADD, DMODEL);

  const int Mrows = NBATCH * SEQ;  // 4096
  // stage 2: QKV projections with fused RoPE / transpose epilogues
  mqa_proj_kernel<<<dim3(NQH * HEADD / 256, Mrows / 64), 256, 0, stream>>>(
      x_h, Wq_t, Mrows, NQH * HEADD, DMODEL, cosp, sinp, q_h, nullptr, 1, NQH);
  mqa_proj_kernel<<<dim3(NKVH * HEADD / 256, Mrows / 64), 256, 0, stream>>>(
      x_h, Wk_t, Mrows, NKVH * HEADD, DMODEL, cosp, sinp, k_h, nullptr, 2, NKVH);
  mqa_proj_kernel<<<dim3(NKVH * HEADD / 256, Mrows / 64), 256, 0, stream>>>(
      x_h, Wv_t, Mrows, NKVH * HEADD, DMODEL, nullptr, nullptr, vT_h, nullptr, 3, NKVH);

  // stage 3: flash attention (wave = 32 q rows, block = 256 rows)
  mqa_attn_kernel<<<dim3(SEQ / 256, NQH, NBATCH), 256, 0, stream>>>(q_h, k_h, vT_h, attn_h);

  // stage 4: output projection (fp32 result)
  mqa_proj_kernel<<<dim3(DMODEL / 256, Mrows / 64), 256, 0, stream>>>(
      attn_h, Wo_t, Mrows, DMODEL, NQH * HEADD, nullptr, nullptr, nullptr, out, 0, 0);
}
